// BitSSM_70720931496895
// MI455X (gfx1250) — matrix operand order repaired and checked by reading, not compiled, not run on hardware
//
#include <hip/hip_runtime.h>
#include <hip/hip_bf16.h>

typedef __attribute__((ext_vector_type(16))) _Float16 v16h;
typedef __attribute__((ext_vector_type(8)))  _Float16 v8h;
typedef __attribute__((ext_vector_type(8)))  float    v8f;

#define D_MODEL 1024
#define D_INNER 2048
#define SEQ     4096
#define NTOK    16384   // B*S = 4*4096

// ---------------------------------------------------------------------------
// WMMA fragment loaders (wave32); this code uses contiguous K order for both A and B.
// A (16x32 f16): lanes 0-15 row M=lane hold K=0..15 contiguous;
//                lanes 16-31 same rows, K=16..31. -> two 16B loads
// B (32x16 f16): lanes 0-15 col N=lane hold K=0..15 contiguous;
//                lanes 16-31 same cols, K=16..31. -> one 32B load
// ---------------------------------------------------------------------------
__device__ __forceinline__ v16h ld_a(const _Float16* p) {
  v8h lo = *(const v8h*)(p);
  v8h hi = *(const v8h*)(p + 8);
  v16h f;
#pragma unroll
  for (int i = 0; i < 8; ++i) { f[i] = lo[i]; f[i + 8] = hi[i]; }
  return f;
}

#define WMMA_F16(acc, a, b)                                                   \
  acc = __builtin_amdgcn_wmma_f32_16x16x32_f16(false, a, false, b, (short)0,  \
                                               acc, false, false)

// ---------------------------------------------------------------------------
// Deterministic |w| mean: pass 1 -> per-block partials, pass 2 -> scale slot.
// ---------------------------------------------------------------------------
__global__ __launch_bounds__(256) void reduce_abs_partial(
    const float* __restrict__ src, long long n, float* __restrict__ partial) {
  __shared__ float sm[256];
  float s = 0.f;
  for (long long i = (long long)blockIdx.x * 256 + threadIdx.x; i < n;
       i += (long long)gridDim.x * 256)
    s += fabsf(src[i]);
  sm[threadIdx.x] = s;
  __syncthreads();
  for (int off = 128; off > 0; off >>= 1) {
    if (threadIdx.x < off) sm[threadIdx.x] += sm[threadIdx.x + off];
    __syncthreads();
  }
  if (threadIdx.x == 0) partial[blockIdx.x] = sm[0];
}

__global__ __launch_bounds__(256) void reduce_abs_final(
    const float* __restrict__ partial, int nb, float inv_count,
    float* __restrict__ scale_out) {
  __shared__ float sm[256];
  float s = 0.f;
  for (int i = threadIdx.x; i < nb; i += 256) s += partial[i];
  sm[threadIdx.x] = s;
  __syncthreads();
  for (int off = 128; off > 0; off >>= 1) {
    if (threadIdx.x < off) sm[threadIdx.x] += sm[threadIdx.x + off];
    __syncthreads();
  }
  if (threadIdx.x == 0) scale_out[0] = fmaxf(sm[0] * inv_count, 1e-5f);
}

// Ternary quantize used rows of a weight into f16 (exact in f16).
__global__ __launch_bounds__(256) void quantize_w(
    const float* __restrict__ w, const float* __restrict__ scale_slot,
    _Float16* __restrict__ wq, long long n) {
  long long i = (long long)blockIdx.x * 256 + threadIdx.x;
  if (i >= n) return;
  float inv = 1.f / scale_slot[0];
  float q = rintf(w[i] * inv);
  q = fminf(1.f, fmaxf(-1.f, q));
  wq[i] = (_Float16)q;
}

// Streaming f32 -> f16 convert (keeps GEMM inner loops conversion-free).
__global__ __launch_bounds__(256) void cvt_f16(
    const float* __restrict__ src, _Float16* __restrict__ dst, long long n) {
  long long i = (long long)blockIdx.x * 256 + threadIdx.x;
  if (i < n) dst[i] = (_Float16)src[i];
}

// ---------------------------------------------------------------------------
// Causal depthwise conv (k=4, pad 3) + bias + SiLU, per-batch boundaries.
// ---------------------------------------------------------------------------
__global__ __launch_bounds__(256) void conv_silu(
    const _Float16* __restrict__ xin, const float* __restrict__ wconv,
    const float* __restrict__ bconv, _Float16* __restrict__ xc) {
  long long i = (long long)blockIdx.x * 256 + threadIdx.x;  // over NTOK*D_INNER
  int c = (int)(i & (D_INNER - 1));
  int t = (int)(i >> 11);
  int s = t & (SEQ - 1);
  float acc = bconv[c];
#pragma unroll
  for (int j = 0; j < 4; ++j) {
    int ss = s - 3 + j;
    if (ss >= 0)
      acc += wconv[c * 4 + j] * (float)xin[(size_t)(t - 3 + j) * D_INNER + c];
  }
  float g = 1.f / (1.f + __expf(-acc));
  xc[i] = (_Float16)(acc * g);
}

// ---------------------------------------------------------------------------
// Tiled WMMA GEMM:  out[M,N] = A[M,K] * Wq[N,K]^T * scale + bias (+ epilogue)
// Block = 8 waves (2 M x 4 N); wave tile = 32(M) x 64(N); block tile 64x256.
// K-loop is register double-buffered (2-stage software pipeline).
// MODE 0: store f16 | MODE 1: y = aux * sigmoid(v), store f16 | MODE 2: f32
// ---------------------------------------------------------------------------
template <int MODE, typename OT>
__global__ __launch_bounds__(256) void bitgemm(
    const _Float16* __restrict__ A, int lda, const _Float16* __restrict__ W,
    int K, const float* __restrict__ scale_slot, const float* __restrict__ bias,
    const _Float16* __restrict__ aux, OT* __restrict__ out, int ldo) {
  const int lane = threadIdx.x & 31;
  const int wave = threadIdx.x >> 5;
  const int m0 = blockIdx.y * 64 + (wave & 1) * 32;
  const int n0 = blockIdx.x * 256 + (wave >> 1) * 64;
  const int hi16 = lane >> 4;
  const int l15  = lane & 15;

  const _Float16* ap0 = A + (size_t)(m0 + l15) * lda + hi16 * 16;
  const _Float16* ap1 = ap0 + (size_t)16 * lda;
  const _Float16* bp0 = W + (size_t)(n0 + l15) * K + hi16 * 16;
  const _Float16* bp1 = bp0 + (size_t)16 * K;
  const _Float16* bp2 = bp0 + (size_t)32 * K;
  const _Float16* bp3 = bp0 + (size_t)48 * K;

  v8f cc[2][4];
#pragma unroll
  for (int r = 0; r < 2; ++r)
#pragma unroll
    for (int t = 0; t < 4; ++t) cc[r][t] = (v8f){};

  // ---- stage 0 preload (k = 0) ----
  v16h a0 = ld_a(ap0), a1 = ld_a(ap1);
  v16h b0 = *(const v16h*)(bp0), b1 = *(const v16h*)(bp1);
  v16h b2 = *(const v16h*)(bp2), b3 = *(const v16h*)(bp3);
  v16h A0, A1, B0, B1, B2, B3;

  int k = 0;
  // K is a multiple of 64; pipeline two K-steps per iteration.
  for (; k + 64 < K; k += 64) {
    // load step k+32 while stage-0 data feeds the matrix pipe
    A0 = ld_a(ap0 + k + 32); A1 = ld_a(ap1 + k + 32);
    B0 = *(const v16h*)(bp0 + k + 32); B1 = *(const v16h*)(bp1 + k + 32);
    B2 = *(const v16h*)(bp2 + k + 32); B3 = *(const v16h*)(bp3 + k + 32);
    __builtin_prefetch(ap0 + k + 512, 0, 1);
    __builtin_prefetch(ap1 + k + 512, 0, 1);
    WMMA_F16(cc[0][0], a0, b0); WMMA_F16(cc[0][1], a0, b1);
    WMMA_F16(cc[0][2], a0, b2); WMMA_F16(cc[0][3], a0, b3);
    WMMA_F16(cc[1][0], a1, b0); WMMA_F16(cc[1][1], a1, b1);
    WMMA_F16(cc[1][2], a1, b2); WMMA_F16(cc[1][3], a1, b3);
    // load step k+64 while stage-1 data feeds the matrix pipe
    a0 = ld_a(ap0 + k + 64); a1 = ld_a(ap1 + k + 64);
    b0 = *(const v16h*)(bp0 + k + 64); b1 = *(const v16h*)(bp1 + k + 64);
    b2 = *(const v16h*)(bp2 + k + 64); b3 = *(const v16h*)(bp3 + k + 64);
    WMMA_F16(cc[0][0], A0, B0); WMMA_F16(cc[0][1], A0, B1);
    WMMA_F16(cc[0][2], A0, B2); WMMA_F16(cc[0][3], A0, B3);
    WMMA_F16(cc[1][0], A1, B0); WMMA_F16(cc[1][1], A1, B1);
    WMMA_F16(cc[1][2], A1, B2); WMMA_F16(cc[1][3], A1, B3);
  }
  // tail: k == K-64; steps K-64 (in regs) and K-32 remain
  A0 = ld_a(ap0 + K - 32); A1 = ld_a(ap1 + K - 32);
  B0 = *(const v16h*)(bp0 + K - 32); B1 = *(const v16h*)(bp1 + K - 32);
  B2 = *(const v16h*)(bp2 + K - 32); B3 = *(const v16h*)(bp3 + K - 32);
  WMMA_F16(cc[0][0], a0, b0); WMMA_F16(cc[0][1], a0, b1);
  WMMA_F16(cc[0][2], a0, b2); WMMA_F16(cc[0][3], a0, b3);
  WMMA_F16(cc[1][0], a1, b0); WMMA_F16(cc[1][1], a1, b1);
  WMMA_F16(cc[1][2], a1, b2); WMMA_F16(cc[1][3], a1, b3);
  WMMA_F16(cc[0][0], A0, B0); WMMA_F16(cc[0][1], A0, B1);
  WMMA_F16(cc[0][2], A0, B2); WMMA_F16(cc[0][3], A0, B3);
  WMMA_F16(cc[1][0], A1, B0); WMMA_F16(cc[1][1], A1, B1);
  WMMA_F16(cc[1][2], A1, B2); WMMA_F16(cc[1][3], A1, B3);

  const float scale = scale_slot[0];
#pragma unroll
  for (int r2 = 0; r2 < 2; ++r2) {
    const int mrow = m0 + r2 * 16 + hi16 * 8;  // C/D layout: VGPR r -> M
#pragma unroll
    for (int t = 0; t < 4; ++t) {
      const int n = n0 + t * 16 + l15;         // C/D layout: lane -> N
      const float bv = bias[n];
      v8f acc = cc[r2][t];
#pragma unroll
      for (int r = 0; r < 8; ++r) {
        float v = acc[r] * scale + bv;
        size_t idx = (size_t)(mrow + r) * ldo + n;
        if (MODE == 0) {
          out[idx] = (OT)v;
        } else if (MODE == 1) {
          float g = 1.f / (1.f + __expf(-v));
          out[idx] = (OT)((float)aux[idx] * g);
        } else {
          out[idx] = (OT)v;
        }
      }
    }
  }
}

// ---------------------------------------------------------------------------
extern "C" void kernel_launch(void* const* d_in, const int* in_sizes, int n_in,
                              void* d_out, int out_size, void* d_ws, size_t ws_size,
                              hipStream_t stream) {
  const float* x      = (const float*)d_in[0];  // (4,4096,1024)
  const float* w_in   = (const float*)d_in[1];  // (4096,1024)
  const float* b_in   = (const float*)d_in[2];  // (4096,)
  const float* w_conv = (const float*)d_in[3];  // (2048,1,4)
  const float* b_conv = (const float*)d_in[4];  // (2048,)
  const float* w_x    = (const float*)d_in[5];  // (3088,2048)
  const float* b_x    = (const float*)d_in[6];  // (3088,)
  const float* w_out  = (const float*)d_in[7];  // (1024,2048)
  const float* b_out  = (const float*)d_in[8];  // (1024,)
  float* out = (float*)d_out;                   // (4,4096,1024)

  char* ws = (char*)d_ws;
  float*     scales  = (float*)ws;                                    // [0..2]
  float*     partial = (float*)(ws + 4096);                           // 1024 f
  _Float16*  w_in_q  = (_Float16*)(ws + 8192);                        // 2048x1024
  _Float16*  w_x_q   = w_in_q  + (size_t)D_INNER * D_MODEL;           // 2048x2048
  _Float16*  w_out_q = w_x_q   + (size_t)D_INNER * D_INNER;           // 1024x2048
  _Float16*  xh      = w_out_q + (size_t)D_MODEL * D_INNER;           // 16384x1024
  _Float16*  xinner  = xh      + (size_t)NTOK * D_MODEL;              // 16384x2048
  _Float16*  xch     = xinner  + (size_t)NTOK * D_INNER;              // 16384x2048
  _Float16*  yh      = xinner;  // x_inner dead after conv -> reuse for y

  // --- weight scales (mean|W| over FULL matrices, as in reference) ---------
  const long long n_in_full  = 4096LL * 1024;
  const long long n_x_full   = 3088LL * 2048;
  const long long n_out_full = 1024LL * 2048;
  reduce_abs_partial<<<1024, 256, 0, stream>>>(w_in, n_in_full, partial);
  reduce_abs_final<<<1, 256, 0, stream>>>(partial, 1024, 1.f / (float)n_in_full, scales + 0);
  reduce_abs_partial<<<1024, 256, 0, stream>>>(w_x, n_x_full, partial);
  reduce_abs_final<<<1, 256, 0, stream>>>(partial, 1024, 1.f / (float)n_x_full, scales + 1);
  reduce_abs_partial<<<1024, 256, 0, stream>>>(w_out, n_out_full, partial);
  reduce_abs_final<<<1, 256, 0, stream>>>(partial, 1024, 1.f / (float)n_out_full, scales + 2);

  // --- ternary-quantize used rows to f16 -----------------------------------
  const long long nq_in  = (long long)D_INNER * D_MODEL;  // rows 0..2047 of w_in
  const long long nq_x   = (long long)D_INNER * D_INNER;  // rows 0..2047 of w_x
  const long long nq_out = (long long)D_MODEL * D_INNER;
  quantize_w<<<(int)((nq_in  + 255) / 256), 256, 0, stream>>>(w_in,  scales + 0, w_in_q,  nq_in);
  quantize_w<<<(int)((nq_x   + 255) / 256), 256, 0, stream>>>(w_x,   scales + 1, w_x_q,   nq_x);
  quantize_w<<<(int)((nq_out + 255) / 256), 256, 0, stream>>>(w_out, scales + 2, w_out_q, nq_out);

  // --- x -> f16 once (keeps all GEMM inner loops pure f16) -----------------
  const long long n_x_elems = (long long)NTOK * D_MODEL;
  cvt_f16<<<(int)((n_x_elems + 255) / 256), 256, 0, stream>>>(x, xh, n_x_elems);

  // --- GEMM1: x_inner = xh @ w_in_q[0:2048]^T * s0 + b_in  (f16 out) -------
  bitgemm<0, _Float16><<<dim3(D_INNER / 256, NTOK / 64), 256, 0, stream>>>(
      xh, D_MODEL, w_in_q, D_MODEL, scales + 0, b_in, nullptr, xinner, D_INNER);

  // --- conv + bias + SiLU --------------------------------------------------
  conv_silu<<<(int)(((long long)NTOK * D_INNER) / 256), 256, 0, stream>>>(
      xinner, w_conv, b_conv, xch);

  // --- GEMM2 fused gate: y = xch * sigmoid(xch @ w_x_q^T * s1 + b_x) -------
  bitgemm<1, _Float16><<<dim3(D_INNER / 256, NTOK / 64), 256, 0, stream>>>(
      xch, D_INNER, w_x_q, D_INNER, scales + 1, b_x, xch, yh, D_INNER);

  // --- GEMM3: out = y @ w_out_q^T * s2 + b_out  (f32 out) ------------------
  bitgemm<2, float><<<dim3(D_MODEL / 256, NTOK / 64), 256, 0, stream>>>(
      yh, D_INNER, w_out_q, D_INNER, scales + 2, b_out, nullptr, out, D_MODEL);
}
